// GraphSAGE_17944373363178
// MI455X (gfx1250) — compile-verified, matrix operand order checked by reading, not run on hardware
//
#include <hip/hip_runtime.h>
#include <hip/hip_bf16.h>

#define N_NODES 100000
#define N_EDGES 1600000
#define D_H 128
#define D_OUT 64
#define MTILES 5                      // 5 x 16 = 80 rows per block; 100000/80 = 1250 exact

typedef __attribute__((ext_vector_type(16))) __bf16 v16bf;
typedef __attribute__((ext_vector_type(8)))  __bf16 v8bf;
typedef __attribute__((ext_vector_type(8)))  float  v8f;

// ---------------------------------------------------------------------------
// Fragment loaders (ISA 7.12.2 layouts, wave32)
// A-matrix 16x32 bf16: lane&15 = row M; lane>>4 selects K-half.
//   elem e<8  -> K = k0 + h*8 + e        (contiguous 8)
//   elem e>=8 -> K = k0 + 16 + h*8 + e-8 (contiguous 8)
// ---------------------------------------------------------------------------
__device__ __forceinline__ v16bf load_a_frag(const float* __restrict__ base,
                                             int row0, int k0, int lane, float scale) {
    const int m = lane & 15;
    const int h = lane >> 4;
    const float* rp = base + (size_t)(row0 + m) * D_H + k0 + h * 8;
    float4 f0 = *(const float4*)(rp);
    float4 f1 = *(const float4*)(rp + 4);
    float4 f2 = *(const float4*)(rp + 16);
    float4 f3 = *(const float4*)(rp + 20);
    v16bf a;
    a[0]  = (__bf16)(f0.x * scale); a[1]  = (__bf16)(f0.y * scale);
    a[2]  = (__bf16)(f0.z * scale); a[3]  = (__bf16)(f0.w * scale);
    a[4]  = (__bf16)(f1.x * scale); a[5]  = (__bf16)(f1.y * scale);
    a[6]  = (__bf16)(f1.z * scale); a[7]  = (__bf16)(f1.w * scale);
    a[8]  = (__bf16)(f2.x * scale); a[9]  = (__bf16)(f2.y * scale);
    a[10] = (__bf16)(f2.z * scale); a[11] = (__bf16)(f2.w * scale);
    a[12] = (__bf16)(f3.x * scale); a[13] = (__bf16)(f3.y * scale);
    a[14] = (__bf16)(f3.z * scale); a[15] = (__bf16)(f3.w * scale);
    return a;
}

// B-matrix 32x16 bf16 from LDS holding W transposed: WT[n*128 + k].
// lane&15 = col N; K mapping mirrors A -> two contiguous 16B ds_load_b128.
__device__ __forceinline__ v16bf load_b_frag_lds(const __bf16* WT,
                                                 int col0, int k0, int lane) {
    const int n = lane & 15;
    const int h = lane >> 4;
    const __bf16* bp = WT + (size_t)(col0 + n) * D_H + k0 + h * 8;
    v8bf lo = *(const v8bf*)(bp);
    v8bf hi = *(const v8bf*)(bp + 16);
    v16bf b;
#pragma unroll
    for (int e = 0; e < 8; ++e) { b[e] = lo[e]; b[e + 8] = hi[e]; }
    return b;
}

// ---------------------------------------------------------------------------
// Utility kernels
// ---------------------------------------------------------------------------
__global__ void zero4_kernel(float* __restrict__ p, long long n4) {
    long long i = (long long)blockIdx.x * blockDim.x + threadIdx.x;
    if (i < n4) ((float4*)p)[i] = make_float4(0.f, 0.f, 0.f, 0.f);
}

__global__ void degree_kernel(const long long* __restrict__ dst, float* __restrict__ deg) {
    int e = blockIdx.x * blockDim.x + threadIdx.x;
    if (e < N_EDGES) unsafeAtomicAdd(&deg[(int)dst[e]], 1.0f);
}

__global__ void invdeg_kernel(float* __restrict__ deg) {
    int i = blockIdx.x * blockDim.x + threadIdx.x;
    if (i < N_NODES) deg[i] = 1.0f / fmaxf(deg[i], 1.0f);
}

// One wave per edge: 32 lanes x float4 = full 128-feature row, coalesced 512B,
// hardware fp32 atomics (global_atomic_add_f32). Gather side is L2-resident.
__global__ void aggregate_kernel(const float* __restrict__ xin,
                                 const long long* __restrict__ src,
                                 const long long* __restrict__ dst,
                                 float* __restrict__ agg) {
    long long t = (long long)blockIdx.x * blockDim.x + threadIdx.x;
    int e    = (int)(t >> 5);
    int lane = (int)(t & 31);
    if (e >= N_EDGES) return;
    int s = (int)src[e];
    int d = (int)dst[e];
    float4 v = ((const float4*)(xin + (size_t)s * D_H))[lane];
    float* ap = agg + (size_t)d * D_H + lane * 4;
    unsafeAtomicAdd(ap + 0, v.x);
    unsafeAtomicAdd(ap + 1, v.y);
    unsafeAtomicAdd(ap + 2, v.z);
    unsafeAtomicAdd(ap + 3, v.w);
}

// ---------------------------------------------------------------------------
// Fused SAGE layer: out = relu( (agg * inv_deg) @ Wl + x @ Wr + bl )
// 256 threads = 8 waves; wave w -> cols [16w,16w+16); block -> 80 rows (5 M-tiles).
// B fragments hoisted per k-step and reused across all 5 M-tiles.
// ---------------------------------------------------------------------------
__global__ void __launch_bounds__(256)
sage_layer_kernel(const float* __restrict__ agg, const float* __restrict__ invdeg,
                  const float* __restrict__ xin,
                  const float* __restrict__ Wl, const float* __restrict__ bl,
                  const float* __restrict__ Wr,
                  float* __restrict__ out) {
    __shared__ __align__(16) __bf16 WlT[D_H * D_H];
    __shared__ __align__(16) __bf16 WrT[D_H * D_H];

    const int tid = threadIdx.x;
    for (int i = tid; i < D_H * D_H; i += 256) {
        int k = i >> 7, n = i & 127;
        WlT[n * D_H + k] = (__bf16)Wl[i];
        WrT[n * D_H + k] = (__bf16)Wr[i];
    }
    __syncthreads();

    const int row0 = blockIdx.x * (16 * MTILES);
    const int lane = tid & 31;
    const int col0 = (tid >> 5) * 16;

    float scale[MTILES];
#pragma unroll
    for (int t = 0; t < MTILES; ++t)
        scale[t] = invdeg[row0 + t * 16 + (lane & 15)];

    v8f c[MTILES];
#pragma unroll
    for (int t = 0; t < MTILES; ++t) c[t] = (v8f){};

#pragma unroll
    for (int k0 = 0; k0 < D_H; k0 += 32) {
        v16bf bL = load_b_frag_lds(WlT, col0, k0, lane);
        v16bf bR = load_b_frag_lds(WrT, col0, k0, lane);
#pragma unroll
        for (int t = 0; t < MTILES; ++t) {
            v16bf aM = load_a_frag(agg, row0 + t * 16, k0, lane, scale[t]);
            c[t] = __builtin_amdgcn_wmma_f32_16x16x32_bf16(false, aM, false, bL,
                                                           (short)0, c[t], false, false);
            v16bf aX = load_a_frag(xin, row0 + t * 16, k0, lane, 1.0f);
            c[t] = __builtin_amdgcn_wmma_f32_16x16x32_bf16(false, aX, false, bR,
                                                           (short)0, c[t], false, false);
        }
    }

    // C/D layout: VGPR r -> M = r + 8*(lane>=16), N = lane&15
    const int n  = lane & 15;
    const int mb = (lane >> 4) * 8;
    const float bias = bl[col0 + n];
#pragma unroll
    for (int t = 0; t < MTILES; ++t) {
#pragma unroll
        for (int r = 0; r < 8; ++r) {
            float v = fmaxf(c[t][r] + bias, 0.0f);
            out[(size_t)(row0 + t * 16 + mb + r) * D_H + col0 + n] = v;
        }
    }
}

// Final projection: out[N,64] = h @ W + b   (128 threads = 4 waves, 5 M-tiles)
__global__ void __launch_bounds__(128)
final_gemm_kernel(const float* __restrict__ h, const float* __restrict__ W,
                  const float* __restrict__ b, float* __restrict__ out) {
    __shared__ __align__(16) __bf16 WT[D_OUT * D_H];
    const int tid = threadIdx.x;
    for (int i = tid; i < D_H * D_OUT; i += 128) {
        int k = i / D_OUT, n = i % D_OUT;
        WT[n * D_H + k] = (__bf16)W[i];
    }
    __syncthreads();

    const int row0 = blockIdx.x * (16 * MTILES);
    const int lane = tid & 31;
    const int col0 = (tid >> 5) * 16;

    v8f c[MTILES];
#pragma unroll
    for (int t = 0; t < MTILES; ++t) c[t] = (v8f){};

#pragma unroll
    for (int k0 = 0; k0 < D_H; k0 += 32) {
        v16bf bb = load_b_frag_lds(WT, col0, k0, lane);
#pragma unroll
        for (int t = 0; t < MTILES; ++t) {
            v16bf a = load_a_frag(h, row0 + t * 16, k0, lane, 1.0f);
            c[t] = __builtin_amdgcn_wmma_f32_16x16x32_bf16(false, a, false, bb,
                                                           (short)0, c[t], false, false);
        }
    }
    const int n  = lane & 15;
    const int mb = (lane >> 4) * 8;
    const float bias = b[col0 + n];
#pragma unroll
    for (int t = 0; t < MTILES; ++t)
#pragma unroll
        for (int r = 0; r < 8; ++r)
            out[(size_t)(row0 + t * 16 + mb + r) * D_OUT + col0 + n] = c[t][r] + bias;
}

// ---------------------------------------------------------------------------
extern "C" void kernel_launch(void* const* d_in, const int* in_sizes, int n_in,
                              void* d_out, int out_size, void* d_ws, size_t ws_size,
                              hipStream_t stream) {
    (void)in_sizes; (void)n_in; (void)out_size; (void)ws_size;

    const float*     x   = (const float*)d_in[0];
    const long long* ei  = (const long long*)d_in[1];
    const long long* src = ei;
    const long long* dst = ei + N_EDGES;
    const float* Wl0 = (const float*)d_in[2];
    const float* bl0 = (const float*)d_in[3];
    const float* Wr0 = (const float*)d_in[4];
    const float* Wl1 = (const float*)d_in[5];
    const float* bl1 = (const float*)d_in[6];
    const float* Wr1 = (const float*)d_in[7];
    const float* Wl2 = (const float*)d_in[8];
    const float* bl2 = (const float*)d_in[9];
    const float* Wr2 = (const float*)d_in[10];
    const float* W   = (const float*)d_in[11];
    const float* b   = (const float*)d_in[12];

    float* ws  = (float*)d_ws;
    float* deg = ws;                                 // N (becomes inv_deg)
    float* agg = ws + N_NODES;                       // N*128
    float* h0  = agg + (size_t)N_NODES * D_H;        // N*128
    float* h1  = h0  + (size_t)N_NODES * D_H;        // N*128
    float* out = (float*)d_out;                      // N*64

    const long long aggN4   = (long long)N_NODES * D_H / 4;
    const int zeroAggBlk    = (int)((aggN4 + 255) / 256);
    const long long aggThr  = (long long)N_EDGES * 32;
    const int aggBlk        = (int)((aggThr + 255) / 256);
    const int gemmBlk       = N_NODES / (16 * MTILES);   // 1250 (exact)

    // degree -> inv_deg
    zero4_kernel<<<(N_NODES / 4 + 255) / 256, 256, 0, stream>>>(deg, N_NODES / 4);
    degree_kernel<<<(N_EDGES + 255) / 256, 256, 0, stream>>>(dst, deg);
    invdeg_kernel<<<(N_NODES + 255) / 256, 256, 0, stream>>>(deg);

    // layer 0: x -> h0
    zero4_kernel<<<zeroAggBlk, 256, 0, stream>>>(agg, aggN4);
    aggregate_kernel<<<aggBlk, 256, 0, stream>>>(x, src, dst, agg);
    sage_layer_kernel<<<gemmBlk, 256, 0, stream>>>(agg, deg, x, Wl0, bl0, Wr0, h0);

    // layer 1: h0 -> h1
    zero4_kernel<<<zeroAggBlk, 256, 0, stream>>>(agg, aggN4);
    aggregate_kernel<<<aggBlk, 256, 0, stream>>>(h0, src, dst, agg);
    sage_layer_kernel<<<gemmBlk, 256, 0, stream>>>(agg, deg, h0, Wl1, bl1, Wr1, h1);

    // layer 2: h1 -> h0
    zero4_kernel<<<zeroAggBlk, 256, 0, stream>>>(agg, aggN4);
    aggregate_kernel<<<aggBlk, 256, 0, stream>>>(h1, src, dst, agg);
    sage_layer_kernel<<<gemmBlk, 256, 0, stream>>>(agg, deg, h1, Wl2, bl2, Wr2, h0);

    // final projection
    final_gemm_kernel<<<gemmBlk, 128, 0, stream>>>(h0, W, b, out);
}
